// TinyLLM_31748398252375
// MI455X (gfx1250) — matrix operand order repaired and checked by reading, not compiled, not run on hardware
//
#include <hip/hip_runtime.h>

// ---------------------------------------------------------------------------
// TinyLLM decode step for MI455X (gfx1250, wave32, WMMA).
// Bandwidth-bound: ~710MB traffic (attn_mem copy dominates) -> ~30us @23.3TB/s.
// Attention k/v projections folded through algebraically; WMMA f16/f32-acc
// used for the LSTM gate GEMMs and the 64x50257x128 logits GEMM.
// ---------------------------------------------------------------------------

#define VOCAB  50257
#define HDIM   128
#define NLAYER 4
#define SEQ    8192
#define BATCH  64
#define CHUNK  512
#define NCHUNK (SEQ / CHUNK)   // 16

typedef __attribute__((ext_vector_type(16))) _Float16 v16h;
typedef __attribute__((ext_vector_type(8)))  float    v8f;

// A-matrix (16x32 f16) per-lane K pattern: lanes0-15 hold K{0..7,16..23},
// lanes16-31 hold K{8..15,24..31}  (ISA 7.12.2, 16-bit A 16x32)
__device__ __forceinline__ int kA(int lane, int j) {
  return ((lane & 16) ? 8 : 0) + j + ((j >= 8) ? 8 : 0);
}
// B-matrix (32x16 f16) per-lane K pattern: same as A, lanes0-15 K{0..7,16..23},
// lanes16-31 K{8..15,24..31}
__device__ __forceinline__ int kB(int lane, int j) {
  return ((lane & 16) ? 8 : 0) + j + ((j >= 8) ? 8 : 0);
}

__device__ __forceinline__ v8f wmma_f16(v16h a, v16h b, v8f c) {
  return __builtin_amdgcn_wmma_f32_16x16x32_f16(
      /*neg_a=*/false, a, /*neg_b=*/false, b,
      /*c_mod=*/(short)0, c, /*reuse_a=*/false, /*reuse_b=*/false);
}

// Load A fragment (16 rows x 32 K) from an f16 LDS tile with row stride HDIM
__device__ __forceinline__ v16h load_a_lds(const _Float16* X, int rowbase,
                                           int kbase, int lane) {
  v16h a;
  const int r = rowbase + (lane & 15);
#pragma unroll
  for (int j = 0; j < 16; ++j) a[j] = X[r * HDIM + kbase + kA(lane, j)];
  return a;
}

// Load B fragment (32 K x 16 cols) from a row-major f32 weight [K, ldn]
__device__ __forceinline__ v16h load_b_glb(const float* W, int ldn, int kbase,
                                           int nbase, int lane) {
  v16h b;
  const int n = nbase + (lane & 15);
#pragma unroll
  for (int j = 0; j < 16; ++j)
    b[j] = (_Float16)W[(size_t)(kbase + kB(lane, j)) * ldn + n];
  return b;
}

__device__ __forceinline__ float sigmoidf_(float x) {
  return 1.0f / (1.0f + expf(-x));
}

// ---------------------------------------------------------------------------
// K1: embedding + 4 LSTM layers. Grid = 4 blocks x 16 batches, 8 waves/block,
// each wave owns one 16-wide slice of the hidden dim. Gates via WMMA f16.
// ---------------------------------------------------------------------------
__global__ void __launch_bounds__(256)
k_lstm(const int* __restrict__ token_id, const float* __restrict__ c0,
       const float* __restrict__ h0, const float* __restrict__ emb,
       const float* __restrict__ Wi, const float* __restrict__ Uh,
       const float* __restrict__ bgate, float* __restrict__ newc,
       float* __restrict__ newh, float* __restrict__ ws_x) {
  __shared__ _Float16 xs[16 * HDIM];  // current layer input (16 batches)
  __shared__ _Float16 hs[16 * HDIM];  // h0[l] for this layer
  const int tid  = threadIdx.x;
  const int lane = tid & 31;
  const int wav  = tid >> 5;          // 0..7 -> hidden-dim tile
  const int bb   = blockIdx.x * 16;   // global batch base

  for (int i = tid; i < 16 * HDIM; i += 256) {
    const int b = i >> 7, e = i & 127;
    xs[i] = (_Float16)emb[(size_t)token_id[bb + b] * HDIM + e];
    hs[i] = (_Float16)h0[(size_t)(bb + b) * HDIM + e];
  }
  __syncthreads();

  const int jn  = wav * 16;           // offset inside each 128-wide gate block
  const int col = lane & 15;
  const int j   = jn + col;

  for (int l = 0; l < NLAYER; ++l) {
    const float* Wl = Wi + (size_t)l * HDIM * 4 * HDIM;
    const float* Ul = Uh + (size_t)l * HDIM * 4 * HDIM;
    const float* bl = bgate + l * 4 * HDIM;
    v8f ai = {}, af = {}, ag = {}, ao = {};
#pragma unroll
    for (int kt = 0; kt < 4; ++kt) {  // x @ Wi
      const v16h a = load_a_lds(xs, 0, kt * 32, lane);
      ai = wmma_f16(a, load_b_glb(Wl, 4 * HDIM, kt * 32, 0 * HDIM + jn, lane), ai);
      af = wmma_f16(a, load_b_glb(Wl, 4 * HDIM, kt * 32, 1 * HDIM + jn, lane), af);
      ag = wmma_f16(a, load_b_glb(Wl, 4 * HDIM, kt * 32, 2 * HDIM + jn, lane), ag);
      ao = wmma_f16(a, load_b_glb(Wl, 4 * HDIM, kt * 32, 3 * HDIM + jn, lane), ao);
    }
#pragma unroll
    for (int kt = 0; kt < 4; ++kt) {  // h0[l] @ Uh
      const v16h a = load_a_lds(hs, 0, kt * 32, lane);
      ai = wmma_f16(a, load_b_glb(Ul, 4 * HDIM, kt * 32, 0 * HDIM + jn, lane), ai);
      af = wmma_f16(a, load_b_glb(Ul, 4 * HDIM, kt * 32, 1 * HDIM + jn, lane), af);
      ag = wmma_f16(a, load_b_glb(Ul, 4 * HDIM, kt * 32, 2 * HDIM + jn, lane), ag);
      ao = wmma_f16(a, load_b_glb(Ul, 4 * HDIM, kt * 32, 3 * HDIM + jn, lane), ao);
    }
    __syncthreads();  // all waves done reading xs/hs

    // elementwise LSTM directly on accumulator fragments
#pragma unroll
    for (int r = 0; r < 8; ++r) {
      const int row = r + ((lane & 16) ? 8 : 0);
      const int gb  = bb + row;
      const float gi = ai[r] + bl[0 * HDIM + j];
      const float gf = af[r] + bl[1 * HDIM + j];
      const float gg = ag[r] + bl[2 * HDIM + j];
      const float go = ao[r] + bl[3 * HDIM + j];
      const size_t ci = ((size_t)l * BATCH + gb) * HDIM + j;
      const float cnew = sigmoidf_(gf) * c0[ci] + sigmoidf_(gi) * tanhf(gg);
      const float hnew = sigmoidf_(go) * tanhf(cnew);
      newc[ci] = cnew;
      newh[ci] = hnew;
      xs[row * HDIM + j] = (_Float16)hnew;  // input for next layer
      if (l == NLAYER - 1) ws_x[(size_t)gb * HDIM + j] = hnew;
    }
    if (l < NLAYER - 1) {
      for (int i = tid; i < 16 * HDIM; i += 256) {
        const int b = i >> 7, e = i & 127;
        hs[i] = (_Float16)h0[((size_t)(l + 1) * BATCH + bb + b) * HDIM + e];
      }
    }
    __syncthreads();
  }
}

// ---------------------------------------------------------------------------
// K2: q = x@Wq+bq, then fold Wk into the query: qk[b]=q[b]@Wk^T, qb[b]=bk.q[b]
// ---------------------------------------------------------------------------
__global__ void __launch_bounds__(256)
k_qk(const float* __restrict__ ws_x, const float* __restrict__ Wq,
     const float* __restrict__ bq, const float* __restrict__ Wk,
     const float* __restrict__ bk, float* __restrict__ qk,
     float* __restrict__ qb) {
  __shared__ float qs[BATCH * HDIM];
  const int tid = threadIdx.x;
  for (int idx = tid; idx < BATCH * HDIM; idx += 256) {
    const int b = idx >> 7, h = idx & 127;
    float s = bq[h];
    for (int e = 0; e < HDIM; ++e) s += ws_x[b * HDIM + e] * Wq[e * HDIM + h];
    qs[idx] = s;
  }
  __syncthreads();
  for (int idx = tid; idx < BATCH * HDIM; idx += 256) {
    const int b = idx >> 7, e = idx & 127;
    float s = 0.f;
    for (int h = 0; h < HDIM; ++h) s += Wk[e * HDIM + h] * qs[b * HDIM + h];
    qk[idx] = s;
  }
  if (tid < BATCH) {
    float s = 0.f;
    for (int h = 0; h < HDIM; ++h) s += bk[h] * qs[tid * HDIM + h];
    qb[tid] = s;
  }
}

// ---------------------------------------------------------------------------
// K3: fused attn_mem copy (with x scattered at row `step`) + score dots.
// One wave per 512B row; 8 waves read 4KB contiguous per iteration.
// ---------------------------------------------------------------------------
__global__ void __launch_bounds__(256)
k_copy_score(const float* __restrict__ mem, const float* __restrict__ ws_x,
             const float* __restrict__ qk, const float* __restrict__ qb,
             const int* __restrict__ step_p, float* __restrict__ omem,
             float* __restrict__ scores) {
  const int step = *step_p;
  const int b    = blockIdx.x >> 7;     // SEQ/64 = 128 groups per batch
  const int grp  = blockIdx.x & 127;
  const int lane = threadIdx.x & 31;
  const int wav  = threadIdx.x >> 5;
  const float4 qv  = ((const float4*)(qk + b * HDIM))[lane];
  const float  qbb = qb[b];
#pragma unroll
  for (int i = 0; i < 8; ++i) {
    const int s = grp * 64 + i * 8 + wav;
    const size_t off = ((size_t)b * SEQ + s) * HDIM;
    float4 v;
    if (s == step) v = ((const float4*)(ws_x + b * HDIM))[lane];
    else           v = ((const float4*)(mem + off))[lane];
    ((float4*)(omem + off))[lane] = v;
    if (s <= step) {
      float d = v.x * qv.x + v.y * qv.y + v.z * qv.z + v.w * qv.w;
#pragma unroll
      for (int o = 16; o > 0; o >>= 1) d += __shfl_xor(d, o, 32);
      if (lane == 0)
        scores[(size_t)b * SEQ + s] = (d + qbb) * 0.08838834764831845f; // 1/sqrt(128)
    }
  }
}

// ---------------------------------------------------------------------------
// K4: per-batch softmax stats (max, 1/sum of exp)
// ---------------------------------------------------------------------------
__global__ void __launch_bounds__(256)
k_stats(const float* __restrict__ scores, const int* __restrict__ step_p,
        float* __restrict__ mv, float* __restrict__ isv) {
  const int step = *step_p;
  const int b    = blockIdx.x;
  __shared__ float red[256];
  const int tid = threadIdx.x;
  float m = -3.0e38f;
  for (int s = tid; s <= step; s += 256) m = fmaxf(m, scores[(size_t)b * SEQ + s]);
  red[tid] = m;
  __syncthreads();
  for (int o = 128; o > 0; o >>= 1) {
    if (tid < o) red[tid] = fmaxf(red[tid], red[tid + o]);
    __syncthreads();
  }
  m = red[0];
  __syncthreads();
  float ss = 0.f;
  for (int s = tid; s <= step; s += 256) ss += expf(scores[(size_t)b * SEQ + s] - m);
  red[tid] = ss;
  __syncthreads();
  for (int o = 128; o > 0; o >>= 1) {
    if (tid < o) red[tid] += red[tid + o];
    __syncthreads();
  }
  if (tid == 0) { mv[b] = m; isv[b] = 1.f / red[0]; }
}

// ---------------------------------------------------------------------------
// K5: chunked softmax-weighted row sum, deterministic partials (no atomics)
// ---------------------------------------------------------------------------
__global__ void __launch_bounds__(128)
k_wsum(const float* __restrict__ omem, const float* __restrict__ scores,
       const int* __restrict__ step_p, const float* __restrict__ mv,
       const float* __restrict__ isv, float* __restrict__ partial) {
  const int step = *step_p;
  const int b  = blockIdx.x & (BATCH - 1);
  const int c  = blockIdx.x / BATCH;
  const int s0 = c * CHUNK;
  if (s0 > step) return;  // uniform early-exit; finalize never reads these
  __shared__ float w[CHUNK];
  const int tid = threadIdx.x;
  const float m = mv[b], inv = isv[b];
  for (int i = tid; i < CHUNK; i += 128) {
    const int s = s0 + i;
    w[i] = (s <= step) ? expf(scores[(size_t)b * SEQ + s] - m) * inv : 0.f;
  }
  __syncthreads();
  float acc = 0.f;
  const float* base = omem + ((size_t)b * SEQ + s0) * HDIM + tid;
  const int smax = (step - s0 + 1 < CHUNK) ? (step - s0 + 1) : CHUNK;
  for (int i = 0; i < smax; ++i) acc += w[i] * base[(size_t)i * HDIM];
  partial[((size_t)c * BATCH + b) * HDIM + tid] = acc;
}

// ---------------------------------------------------------------------------
// K6: combine partials -> ctx@Wv -> tanh([x|ctx]@Wmix) -> LayerNorm
// ---------------------------------------------------------------------------
__global__ void __launch_bounds__(256)
k_mix(const float* __restrict__ partial, const int* __restrict__ step_p,
      const float* __restrict__ ws_x, const float* __restrict__ Wv,
      const float* __restrict__ bv, const float* __restrict__ Wmix,
      const float* __restrict__ bmix, const float* __restrict__ lns,
      const float* __restrict__ lnb, float* __restrict__ ws_ctx,
      float* __restrict__ hln) {
  const int step = *step_p;
  const int nch  = step / CHUNK + 1;
  __shared__ float buf[BATCH * HDIM];  // mctx, later reused for h_mix
  const int tid = threadIdx.x;
  for (int idx = tid; idx < BATCH * HDIM; idx += 256) {
    float a = 0.f;
    for (int c = 0; c < nch; ++c) a += partial[(size_t)c * BATCH * HDIM + idx];
    buf[idx] = a;
  }
  __syncthreads();
  for (int idx = tid; idx < BATCH * HDIM; idx += 256) {
    const int b = idx >> 7, h = idx & 127;
    float s = bv[h];
    for (int e = 0; e < HDIM; ++e) s += buf[b * HDIM + e] * Wv[e * HDIM + h];
    ws_ctx[idx] = s;
  }
  __syncthreads();
  for (int idx = tid; idx < BATCH * HDIM; idx += 256) {
    const int b = idx >> 7, h = idx & 127;
    float s = bmix[h];
    for (int e = 0; e < HDIM; ++e) s += ws_x[b * HDIM + e] * Wmix[e * HDIM + h];
    for (int e = 0; e < HDIM; ++e)
      s += ws_ctx[b * HDIM + e] * Wmix[(HDIM + e) * HDIM + h];
    buf[idx] = tanhf(s);
  }
  __syncthreads();
  const int lane = tid & 31, wav = tid >> 5;  // one wave per row group
  for (int b = wav; b < BATCH; b += 8) {
    float v[4], s1 = 0.f, s2 = 0.f;
#pragma unroll
    for (int t = 0; t < 4; ++t) { v[t] = buf[b * HDIM + lane * 4 + t]; s1 += v[t]; }
#pragma unroll
    for (int o = 16; o > 0; o >>= 1) s1 += __shfl_xor(s1, o, 32);
    const float mu = s1 * (1.f / 128.f);
#pragma unroll
    for (int t = 0; t < 4; ++t) { const float d = v[t] - mu; s2 += d * d; }
#pragma unroll
    for (int o = 16; o > 0; o >>= 1) s2 += __shfl_xor(s2, o, 32);
    const float rs = rsqrtf(s2 * (1.f / 128.f) + 1e-6f);
#pragma unroll
    for (int t = 0; t < 4; ++t) {
      const int h = lane * 4 + t;
      hln[b * HDIM + h] = (v[t] - mu) * rs * lns[h] + lnb[h];
    }
  }
}

// ---------------------------------------------------------------------------
// K7: logits = h_ln[64x128] @ Wout[128xV] + bout, WMMA f16 inputs / f32 acc.
// Block = 4 waves x 16 columns; each wave loads each B fragment once and
// reuses it across the 4 batch tiles.
// ---------------------------------------------------------------------------
__global__ void __launch_bounds__(128)
k_logits(const float* __restrict__ hln, const float* __restrict__ Wout,
         const float* __restrict__ bout, float* __restrict__ logits) {
  __shared__ _Float16 as[BATCH * HDIM];  // 16KB
  const int tid = threadIdx.x, lane = tid & 31, wav = tid >> 5;
  for (int i = tid; i < BATCH * HDIM; i += 128) as[i] = (_Float16)hln[i];
  __syncthreads();

  const int nb = blockIdx.x * 64 + wav * 16;  // column base for this wave
  const int n  = nb + (lane & 15);
  v8f a0 = {}, a1 = {}, a2 = {}, a3 = {};
#pragma unroll
  for (int kt = 0; kt < 4; ++kt) {
    v16h bf;
#pragma unroll
    for (int j = 0; j < 16; ++j) {
      const int k = kt * 32 + kB(lane, j);
      const float wv = (n < VOCAB) ? Wout[(size_t)k * VOCAB + n] : 0.f;
      bf[j] = (_Float16)wv;
    }
    a0 = wmma_f16(load_a_lds(as,  0, kt * 32, lane), bf, a0);
    a1 = wmma_f16(load_a_lds(as, 16, kt * 32, lane), bf, a1);
    a2 = wmma_f16(load_a_lds(as, 32, kt * 32, lane), bf, a2);
    a3 = wmma_f16(load_a_lds(as, 48, kt * 32, lane), bf, a3);
  }
  if (n < VOCAB) {
    const float bo = bout[n];
#pragma unroll
    for (int r = 0; r < 8; ++r) {
      const int row = r + ((lane & 16) ? 8 : 0);
      logits[(size_t)(row +  0) * VOCAB + n] = a0[r] + bo;
      logits[(size_t)(row + 16) * VOCAB + n] = a1[r] + bo;
      logits[(size_t)(row + 32) * VOCAB + n] = a2[r] + bo;
      logits[(size_t)(row + 48) * VOCAB + n] = a3[r] + bo;
    }
  }
}

// ---------------------------------------------------------------------------
extern "C" void kernel_launch(void* const* d_in, const int* in_sizes, int n_in,
                              void* d_out, int out_size, void* d_ws,
                              size_t ws_size, hipStream_t stream) {
  (void)in_sizes; (void)n_in; (void)out_size; (void)ws_size;
  const int*   token_id = (const int*)  d_in[0];
  const float* c0       = (const float*)d_in[1];
  const float* h0       = (const float*)d_in[2];
  const float* attn_mem = (const float*)d_in[3];
  const int*   step_p   = (const int*)  d_in[4];
  const float* emb      = (const float*)d_in[5];
  const float* Wi       = (const float*)d_in[6];
  const float* Uh       = (const float*)d_in[7];
  const float* lb       = (const float*)d_in[8];
  const float* Wq       = (const float*)d_in[9];
  const float* bq       = (const float*)d_in[10];
  const float* Wk       = (const float*)d_in[11];
  const float* bk       = (const float*)d_in[12];
  const float* Wv       = (const float*)d_in[13];
  const float* bv       = (const float*)d_in[14];
  const float* Wmix     = (const float*)d_in[15];
  const float* bmix     = (const float*)d_in[16];
  const float* lns      = (const float*)d_in[17];
  const float* lnb      = (const float*)d_in[18];
  const float* Wout     = (const float*)d_in[19];
  const float* bout     = (const float*)d_in[20];

  float* out    = (float*)d_out;      // [logits | new_c | new_h | attn_mem]
  float* logits = out;
  float* newc   = logits + (size_t)BATCH * VOCAB;
  float* newh   = newc + (size_t)NLAYER * BATCH * HDIM;
  float* omem   = newh + (size_t)NLAYER * BATCH * HDIM;

  float* W       = (float*)d_ws;
  float* ws_x    = W;                        // 8192
  float* qk      = W + 8192;                 // 8192
  float* qb      = W + 16384;                // 64
  float* mv      = W + 16448;                // 64
  float* isv     = W + 16512;                // 64
  float* ws_ctx  = W + 16576;                // 8192
  float* hln     = W + 24768;                // 8192
  float* scores  = W + 32960;                // 64*8192
  float* partial = scores + (size_t)BATCH * SEQ;  // 16*64*128

  k_lstm<<<dim3(4), dim3(256), 0, stream>>>(token_id, c0, h0, emb, Wi, Uh, lb,
                                            newc, newh, ws_x);
  k_qk<<<dim3(1), dim3(256), 0, stream>>>(ws_x, Wq, bq, Wk, bk, qk, qb);
  k_copy_score<<<dim3(BATCH * (SEQ / 64)), dim3(256), 0, stream>>>(
      attn_mem, ws_x, qk, qb, step_p, omem, scores);
  k_stats<<<dim3(BATCH), dim3(256), 0, stream>>>(scores, step_p, mv, isv);
  k_wsum<<<dim3(NCHUNK * BATCH), dim3(128), 0, stream>>>(omem, scores, step_p,
                                                         mv, isv, partial);
  k_mix<<<dim3(1), dim3(256), 0, stream>>>(partial, step_p, ws_x, Wv, bv, Wmix,
                                           bmix, lns, lnb, ws_ctx, hln);
  k_logits<<<dim3((VOCAB + 63) / 64), dim3(128), 0, stream>>>(hln, Wout, bout,
                                                              logits);
}
